// CausalSelfAttention_29515015258562
// MI455X (gfx1250) — matrix operand order repaired and checked by reading, not compiled, not run on hardware
//
#include <hip/hip_runtime.h>
#include <hip/hip_bf16.h>
#include <math.h>

typedef __attribute__((ext_vector_type(16))) __bf16 v16bf;
typedef __attribute__((ext_vector_type(8)))  __bf16 v8bf;
typedef __attribute__((ext_vector_type(8)))  float  v8f;
typedef unsigned short u16v8 __attribute__((ext_vector_type(8)));

union BFrag { v16bf v; v8bf h[2]; };

constexpr int B_ = 2, S_ = 2048, D_ = 1024, H_ = 16, HD_ = 64;
// 1/sqrt(HD) * log2(e), folded into Q so softmax can use exp2 directly.
constexpr float QSCALE_ = 0.125f * 1.4426950408889634f;

__device__ __forceinline__ unsigned short f32_to_bf16(float f) {
    unsigned int u = __float_as_uint(f);
    unsigned int r = (u + 0x7FFFu + ((u >> 16) & 1u)) >> 16;   // RNE
    return (unsigned short)r;
}

// ---------------- Stage 1: f32 -> bf16 convert (8 elem / thread) ----------------
__global__ void cvt_bf16_kernel(const float4* __restrict__ src,
                                u16v8* __restrict__ dst, int n8) {
    int i = blockIdx.x * blockDim.x + threadIdx.x;
    if (i < n8) {
        const float4 a = src[2 * i];
        const float4 b = src[2 * i + 1];
        u16v8 o;
        o[0] = f32_to_bf16(a.x); o[1] = f32_to_bf16(a.y);
        o[2] = f32_to_bf16(a.z); o[3] = f32_to_bf16(a.w);
        o[4] = f32_to_bf16(b.x); o[5] = f32_to_bf16(b.y);
        o[6] = f32_to_bf16(b.z); o[7] = f32_to_bf16(b.w);
        dst[i] = o;
    }
}

// ---------------- Stage 2: fused QKV projection GEMM ----------------
// Y = X (4096x1024) * W^T (1024x1024) + b.  Each wave: 64x64 tile.
// z=0 -> Q*QSCALE [B,H,S,HD], z=1 -> K [B,H,S,HD], z=2 -> V^T [B,H,HD,S]
__global__ __launch_bounds__(256)
void qkv_gemm_kernel(const unsigned short* __restrict__ Xb,
                     const unsigned short* __restrict__ Wqb,
                     const unsigned short* __restrict__ Wkb,
                     const unsigned short* __restrict__ Wvb,
                     const float* __restrict__ bq,
                     const float* __restrict__ bk,
                     const float* __restrict__ bv,
                     unsigned short* __restrict__ Qo,
                     unsigned short* __restrict__ Ko,
                     unsigned short* __restrict__ Vto) {
    const int z = blockIdx.z;
    const unsigned short* W = (z == 0) ? Wqb : (z == 1) ? Wkb : Wvb;
    const float* bias       = (z == 0) ? bq  : (z == 1) ? bk  : bv;
    unsigned short* out     = (z == 0) ? Qo  : (z == 1) ? Ko  : Vto;

    const int lane = threadIdx.x & 31;
    const int wave = threadIdx.x >> 5;
    const int lrow = lane & 15;
    const int hi   = lane >> 4;                    // 0 or 1 (lane half)
    const int base_m = blockIdx.x * 64;
    const int base_n = blockIdx.y * 512 + wave * 64;

    v8f acc[4][4];
#pragma unroll
    for (int mt = 0; mt < 4; ++mt)
#pragma unroll
        for (int nt = 0; nt < 4; ++nt) acc[mt][nt] = (v8f)0.0f;

    for (int kk = 0; kk < D_; kk += 32) {
        BFrag a[4], b[4];
        const int akb = kk + hi * 8;               // A operand K base
#pragma unroll
        for (int mt = 0; mt < 4; ++mt) {
            const unsigned short* rp = Xb + (size_t)(base_m + mt * 16 + lrow) * D_ + akb;
            a[mt].h[0] = *reinterpret_cast<const v8bf*>(rp);       // K = akb..akb+7
            a[mt].h[1] = *reinterpret_cast<const v8bf*>(rp + 16);  // K = akb+16..akb+23
        }
        const int bkh = kk + hi * 8;               // B operand K base
#pragma unroll
        for (int nt = 0; nt < 4; ++nt) {
            const unsigned short* rp = W + (size_t)(base_n + nt * 16 + lrow) * D_ + bkh;
            b[nt].h[0] = *reinterpret_cast<const v8bf*>(rp);       // K = bkh..bkh+7
            b[nt].h[1] = *reinterpret_cast<const v8bf*>(rp + 16);  // K = bkh+16..bkh+23
        }
#pragma unroll
        for (int mt = 0; mt < 4; ++mt)
#pragma unroll
            for (int nt = 0; nt < 4; ++nt)
                acc[mt][nt] = __builtin_amdgcn_wmma_f32_16x16x32_bf16(
                    false, a[mt].v, false, b[nt].v, (short)0, acc[mt][nt], false, false);
    }

    // Epilogue: bias add (+Q scaling), bf16 convert, permuted store
    const float oscale = (z == 0) ? QSCALE_ : 1.0f;
#pragma unroll
    for (int nt = 0; nt < 4; ++nt) {
        const int n  = base_n + nt * 16 + lrow;
        const float bsv = bias[n];
        const int h  = n >> 6;                     // /HD
        const int hd = n & (HD_ - 1);
#pragma unroll
        for (int mt = 0; mt < 4; ++mt) {
            const int m0 = base_m + mt * 16 + hi * 8;    // r=0 row
            const int bb = m0 >> 11;                     // /S (block never straddles batches)
            const int s0 = m0 & (S_ - 1);
            if (z < 2) {
                // [B,H,S,HD]: r varies s -> stride HD in memory, scalar stores
#pragma unroll
                for (int r = 0; r < 8; ++r) {
                    const size_t idx = (((size_t)(bb * H_ + h)) * S_ + (s0 + r)) * HD_ + hd;
                    out[idx] = f32_to_bf16((acc[mt][nt][r] + bsv) * oscale);
                }
            } else {
                // [B,H,HD,S]: r varies s -> contiguous, one 16B store
                u16v8 pk;
#pragma unroll
                for (int r = 0; r < 8; ++r) pk[r] = f32_to_bf16(acc[mt][nt][r] + bsv);
                const size_t idx = (((size_t)(bb * H_ + h)) * HD_ + hd) * S_ + s0;
                *reinterpret_cast<u16v8*>(out + idx) = pk;
            }
        }
    }
}

// ---------------- Stage 3: flash attention (exp2 domain) ----------------
// Each wave: 32 q-rows; key chunks of 32; online softmax; O = P*V via LDS reshape.
__global__ __launch_bounds__(256)
void attn_kernel(const unsigned short* __restrict__ Qb,
                 const unsigned short* __restrict__ Kb,
                 const unsigned short* __restrict__ Vtb,
                 const int* __restrict__ amask,
                 float* __restrict__ out) {
    __shared__ unsigned short plds[8 * 32 * 32];   // 16 KB, per-wave 32x32 P tile

    const int lane = threadIdx.x & 31;
    const int wave = threadIdx.x >> 5;
    const int lrow = lane & 15;
    const int hi   = lane >> 4;
    const int bh   = blockIdx.y;                   // b*H + h
    const int b    = bh >> 4;
    const int h    = bh & (H_ - 1);
    const int q0   = blockIdx.x * 256 + wave * 32;

    const unsigned short* Qh = Qb  + (size_t)bh * S_ * HD_;
    const unsigned short* Kh = Kb  + (size_t)bh * S_ * HD_;
    const unsigned short* Vh = Vtb + (size_t)bh * HD_ * S_;
    const int* am = amask + b * S_;

    // Load Q fragments once: [mtile][kstep over HD]
    BFrag qf[2][2];
#pragma unroll
    for (int mt = 0; mt < 2; ++mt) {
        const unsigned short* rp = Qh + (size_t)(q0 + mt * 16 + lrow) * HD_;
#pragma unroll
        for (int ks = 0; ks < 2; ++ks) {
            const int kb = ks * 32 + hi * 8;
            qf[mt][ks].h[0] = *reinterpret_cast<const v8bf*>(rp + kb);
            qf[mt][ks].h[1] = *reinterpret_cast<const v8bf*>(rp + kb + 16);
        }
    }

    v8f oacc[2][4];
    float mrow[2][8], lsum[2][8];
#pragma unroll
    for (int mt = 0; mt < 2; ++mt) {
#pragma unroll
        for (int nt = 0; nt < 4; ++nt) oacc[mt][nt] = (v8f)0.0f;
#pragma unroll
        for (int r = 0; r < 8; ++r) { mrow[mt][r] = -1.0e30f; lsum[mt][r] = 0.0f; }
    }

    unsigned short* pw = plds + wave * (32 * 32);
    const int kend = q0 + 32;                      // causal bound (keys < kend)

    for (int kc = 0; kc < kend; kc += 32) {
        const bool diag = (kc == q0);              // only diagonal chunk is triangular

        // Prefetch next chunk's K rows and V cachelines into near caches
        if (kc + 32 < kend) {
            __builtin_prefetch(Kh + (size_t)(kc + 32 + lane) * HD_, 0, 1);
            __builtin_prefetch(Vh + (size_t)lane * S_ + kc + 32, 0, 1);
            __builtin_prefetch(Vh + (size_t)(lane + 32) * S_ + kc + 32, 0, 1);
        }

        // ---- S(log2-domain) = (Q*QSCALE) * K^T over HD=64 (2 ksteps) ----
        v8f sacc[2][2];
#pragma unroll
        for (int mt = 0; mt < 2; ++mt)
#pragma unroll
            for (int nt = 0; nt < 2; ++nt) sacc[mt][nt] = (v8f)0.0f;

#pragma unroll
        for (int ks = 0; ks < 2; ++ks) {
            BFrag kf[2];
            const int kh = ks * 32 + hi * 8;
#pragma unroll
            for (int nt = 0; nt < 2; ++nt) {
                const unsigned short* rp = Kh + (size_t)(kc + nt * 16 + lrow) * HD_ + kh;
                kf[nt].h[0] = *reinterpret_cast<const v8bf*>(rp);
                kf[nt].h[1] = *reinterpret_cast<const v8bf*>(rp + 16);
            }
#pragma unroll
            for (int mt = 0; mt < 2; ++mt)
#pragma unroll
                for (int nt = 0; nt < 2; ++nt)
                    sacc[mt][nt] = __builtin_amdgcn_wmma_f32_16x16x32_bf16(
                        false, qf[mt][ks].v, false, kf[nt].v, (short)0, sacc[mt][nt], false, false);
        }

        // ---- pad mask as additive bias; causal cndmask only on diagonal chunk ----
#pragma unroll
        for (int nt = 0; nt < 2; ++nt) {
            const int key = kc + nt * 16 + lrow;
            const float pb = (am[key] == 0) ? 0.0f : -1.0e30f;
            if (diag) {
#pragma unroll
                for (int mt = 0; mt < 2; ++mt)
#pragma unroll
                    for (int r = 0; r < 8; ++r) {
                        const int qrow = q0 + mt * 16 + r + hi * 8;
                        float sv = sacc[mt][nt][r] + pb;
                        if (key > qrow) sv = -1.0e30f;
                        sacc[mt][nt][r] = sv;
                    }
            } else {
#pragma unroll
                for (int mt = 0; mt < 2; ++mt)
#pragma unroll
                    for (int r = 0; r < 8; ++r) sacc[mt][nt][r] += pb;
            }
        }

        // ---- online softmax in exp2 domain (rows live across 16-lane half) ----
#pragma unroll
        for (int mt = 0; mt < 2; ++mt) {
#pragma unroll
            for (int r = 0; r < 8; ++r) {
                float v = fmaxf(sacc[mt][0][r], sacc[mt][1][r]);
#pragma unroll
                for (int off = 8; off >= 1; off >>= 1) v = fmaxf(v, __shfl_xor(v, off, 32));
                const float mnew  = fmaxf(mrow[mt][r], v);
                const float alpha = exp2f(mrow[mt][r] - mnew);
                mrow[mt][r] = mnew;
                const float p0 = exp2f(sacc[mt][0][r] - mnew);
                const float p1 = exp2f(sacc[mt][1][r] - mnew);
                sacc[mt][0][r] = p0; sacc[mt][1][r] = p1;
                float s = p0 + p1;
#pragma unroll
                for (int off = 8; off >= 1; off >>= 1) s += __shfl_xor(s, off, 32);
                lsum[mt][r] = lsum[mt][r] * alpha + s;
#pragma unroll
                for (int nt = 0; nt < 4; ++nt) oacc[mt][nt][r] *= alpha;
            }
        }

        // ---- P: C-layout -> LDS (row-major 32x32 bf16) -> A-operand layout ----
#pragma unroll
        for (int mt = 0; mt < 2; ++mt)
#pragma unroll
            for (int nt = 0; nt < 2; ++nt)
#pragma unroll
                for (int r = 0; r < 8; ++r) {
                    const int ml = mt * 16 + r + hi * 8;
                    const int kl = nt * 16 + lrow;
                    pw[ml * 32 + kl] = f32_to_bf16(sacc[mt][nt][r]);
                }
        asm volatile("s_wait_dscnt 0" ::: "memory");

        BFrag pf[2];
#pragma unroll
        for (int mt = 0; mt < 2; ++mt) {
            const unsigned short* rp = pw + (mt * 16 + lrow) * 32 + hi * 8;
            pf[mt].h[0] = *reinterpret_cast<const v8bf*>(rp);
            pf[mt].h[1] = *reinterpret_cast<const v8bf*>(rp + 16);
        }

        // ---- O += P * V  (V^T rows are contiguous in key) ----
#pragma unroll
        for (int nt = 0; nt < 4; ++nt) {
            const unsigned short* rp = Vh + (size_t)(nt * 16 + lrow) * S_ + kc + hi * 8;
            BFrag vf;
            vf.h[0] = *reinterpret_cast<const v8bf*>(rp);
            vf.h[1] = *reinterpret_cast<const v8bf*>(rp + 16);
#pragma unroll
            for (int mt = 0; mt < 2; ++mt)
                oacc[mt][nt] = __builtin_amdgcn_wmma_f32_16x16x32_bf16(
                    false, pf[mt].v, false, vf.v, (short)0, oacc[mt][nt], false, false);
        }
    }

    // ---- finalize: O / l, store [B,S,H,HD] f32 ----
#pragma unroll
    for (int mt = 0; mt < 2; ++mt) {
#pragma unroll
        for (int r = 0; r < 8; ++r) {
            const float inv = 1.0f / lsum[mt][r];
            const int s = q0 + mt * 16 + r + hi * 8;
            float* orow = out + (((size_t)b * S_ + s) * H_ + h) * HD_;
#pragma unroll
            for (int nt = 0; nt < 4; ++nt) {
                const int hd = nt * 16 + lrow;
                orow[hd] = oacc[mt][nt][r] * inv;
            }
        }
    }
}

// ---------------- launch ----------------
extern "C" void kernel_launch(void* const* d_in, const int* in_sizes, int n_in,
                              void* d_out, int out_size, void* d_ws, size_t ws_size,
                              hipStream_t stream) {
    const float* hs    = (const float*)d_in[0];
    const int*   amask = (const int*)d_in[1];
    const float* Wq    = (const float*)d_in[2];
    const float* bq    = (const float*)d_in[3];
    const float* Wk    = (const float*)d_in[4];
    const float* bk    = (const float*)d_in[5];
    const float* Wv    = (const float*)d_in[6];
    const float* bv    = (const float*)d_in[7];
    float* out = (float*)d_out;

    // Workspace layout (all bf16 as ushort): 38 MB total
    char* ws = (char*)d_ws;
    const size_t MB = 1024ull * 1024ull;
    unsigned short* Xb  = (unsigned short*)(ws);               //  8 MB [4096,1024]
    unsigned short* Wqb = (unsigned short*)(ws + 8 * MB);      //  2 MB
    unsigned short* Wkb = (unsigned short*)(ws + 10 * MB);     //  2 MB
    unsigned short* Wvb = (unsigned short*)(ws + 12 * MB);     //  2 MB
    unsigned short* Qb  = (unsigned short*)(ws + 14 * MB);     //  8 MB [B,H,S,HD], pre-scaled
    unsigned short* Kb  = (unsigned short*)(ws + 22 * MB);     //  8 MB [B,H,S,HD]
    unsigned short* Vtb = (unsigned short*)(ws + 30 * MB);     //  8 MB [B,H,HD,S]

    const int nX8 = (B_ * S_ * D_) / 8;   // 512K vec8 elements
    const int nW8 = (D_ * D_) / 8;        // 128K vec8 elements
    cvt_bf16_kernel<<<(nX8 + 255) / 256, 256, 0, stream>>>((const float4*)hs, (u16v8*)Xb, nX8);
    cvt_bf16_kernel<<<(nW8 + 255) / 256, 256, 0, stream>>>((const float4*)Wq, (u16v8*)Wqb, nW8);
    cvt_bf16_kernel<<<(nW8 + 255) / 256, 256, 0, stream>>>((const float4*)Wk, (u16v8*)Wkb, nW8);
    cvt_bf16_kernel<<<(nW8 + 255) / 256, 256, 0, stream>>>((const float4*)Wv, (u16v8*)Wvb, nW8);

    qkv_gemm_kernel<<<dim3(64, 2, 3), 256, 0, stream>>>(
        Xb, Wqb, Wkb, Wvb, bq, bk, bv, Qb, Kb, Vtb);

    attn_kernel<<<dim3(S_ / 256, B_ * H_), 256, 0, stream>>>(
        Qb, Kb, Vtb, amask, out);
}
